// n_pair_loss_2353642078317
// MI455X (gfx1250) — matrix operand order repaired and checked by reading, not compiled, not run on hardware
//
#include <hip/hip_runtime.h>
#include <math.h>

typedef __attribute__((ext_vector_type(16))) _Float16 v16h;
typedef __attribute__((ext_vector_type(8)))  _Float16 v8h;
typedef __attribute__((ext_vector_type(4)))  _Float16 v4h;
typedef __attribute__((ext_vector_type(8)))  float    v8f;

#define DIM        128
#define KNEG       64
#define A_TOTAL    8192
#define B_TOTAL    16384
#define NPAIR_BLOCKS 512   // 512 blocks * 8 waves = 4096 waves, 2 anchors each
#define L2_BLOCKS    64

__device__ __forceinline__ float wave_sum32(float v) {
#pragma unroll
  for (int m = 16; m > 0; m >>= 1) v += __shfl_xor(v, m, 32);
  return v;
}
__device__ __forceinline__ float wave_max32(float v) {
#pragma unroll
  for (int m = 16; m > 0; m >>= 1) v = fmaxf(v, __shfl_xor(v, m, 32));
  return v;
}

// ---------------------------------------------------------------------------
// One-shot precision split of the embedding table: x = hi(f16) + lo(f16).
// Row layout preserved (16384 x 128 halves per plane), so gathered WMMA
// fragments become raw 16B/32B loads with zero per-use conversion math.
// ---------------------------------------------------------------------------
__global__ __launch_bounds__(256) void split_kernel(
    const float* __restrict__ embed,
    _Float16*    __restrict__ hip_,
    _Float16*    __restrict__ lop_)
{
  const int t = blockIdx.x * 256 + threadIdx.x;      // 4 elements per thread
  float4 x = ((const float4*)embed)[t];
  v4h h, l;
  h[0] = (_Float16)x.x;  l[0] = (_Float16)(x.x - (float)h[0]);
  h[1] = (_Float16)x.y;  l[1] = (_Float16)(x.y - (float)h[1]);
  h[2] = (_Float16)x.z;  l[2] = (_Float16)(x.z - (float)h[2]);
  h[3] = (_Float16)x.w;  l[3] = (_Float16)(x.w - (float)h[3]);
  ((v4h*)hip_)[t] = h;
  ((v4h*)lop_)[t] = l;
}

// ---------------------------------------------------------------------------
// One wave per anchor. inner[64] = N(64x128) . a(128) via split-precision
// f16 WMMA: x = hi + lo, inner ~= nh*ah + nh*al + nl*ah  (~22-bit mantissa).
// PLANES=true : fragments loaded directly from pre-split f16 planes.
// PLANES=false: fragments converted from f32 in-loop (ws too small fallback).
// ---------------------------------------------------------------------------
template <bool PLANES>
__global__ __launch_bounds__(256) void npair_kernel(
    const float*    __restrict__ embed,
    const _Float16* __restrict__ hip_,
    const _Float16* __restrict__ lop_,
    const int*      __restrict__ anc_ind,
    const int*      __restrict__ pos_ind,
    const int*      __restrict__ neg_ind,
    float*          __restrict__ partial)
{
  const int lane      = threadIdx.x & 31;
  const int waveInBlk = threadIdx.x >> 5;
  const int wave      = blockIdx.x * 8 + waveInBlk;   // 0..4095
  const int hi16      = (lane >= 16) ? 1 : 0;
  const int laneM     = lane & 15;

  float local = 0.0f;

#pragma unroll 1
  for (int rep = 0; rep < 2; ++rep) {
    const int i = wave + rep * (NPAIR_BLOCKS * 8);    // anchor id, always < 8192
    const size_t abase = (size_t)anc_ind[i] * DIM;
    const float* __restrict__ arow = embed + abase;
    const float* __restrict__ prow = embed + (size_t)pos_ind[i] * DIM;

    // s = a . p  (f32; each lane handles 4 contiguous elements)
    float4 av = ((const float4*)arow)[lane];
    float4 pv = ((const float4*)prow)[lane];
    float s = wave_sum32(av.x*pv.x + av.y*pv.y + av.z*pv.z + av.w*pv.w);

    // this lane's negative row base offset in each of the 4 M-tiles
    size_t nbase[4];
#pragma unroll
    for (int t = 0; t < 4; ++t)
      nbase[t] = (size_t)neg_ind[i * KNEG + t * 16 + laneM] * DIM;

    v8f acc[4];
#pragma unroll
    for (int t = 0; t < 4; ++t) acc[t] = (v8f){};

#pragma unroll
    for (int c = 0; c < 4; ++c) {                     // D chunks of 32
      // B fragment (anchor vector replicated into all 16 columns):
      // lanes 0-15 hold K = c*32 + 0..15, lanes 16-31 hold K = c*32 + 16..31
      const int bbase = c * 32 + (hi16 ? 16 : 0);
      v16h bh, bl;
      if (PLANES) {
        bh = *(const v16h*)(hip_ + abase + bbase);    // 32B, 32B-aligned
        bl = *(const v16h*)(lop_ + abase + bbase);
      } else {
#pragma unroll
        for (int j = 0; j < 16; ++j) {
          float x = arow[bbase + j];
          _Float16 h = (_Float16)x;
          bh[j] = h;
          bl[j] = (_Float16)(x - (float)h);
        }
      }
#pragma unroll
      for (int t = 0; t < 4; ++t) {                   // 16-negative tiles
        // A fragment (16-bit A 16x32): lane<16 -> K {0..15},
        // lane>=16 -> K {16..31} of row M = lane&15.
        const int off0 = c * 32 + (hi16 ? 16 : 0);
        v16h ah, al;
        if (PLANES) {
          const _Float16* nh = hip_ + nbase[t];
          const _Float16* nl = lop_ + nbase[t];
          v8h h0 = *(const v8h*)(nh + off0);          // 16B, 16B-aligned
          v8h h1 = *(const v8h*)(nh + off0 + 8);
          v8h l0 = *(const v8h*)(nl + off0);
          v8h l1 = *(const v8h*)(nl + off0 + 8);
#pragma unroll
          for (int j = 0; j < 8; ++j) {
            ah[j] = h0[j]; ah[8 + j] = h1[j];
            al[j] = l0[j]; al[8 + j] = l1[j];
          }
        } else {
          const float* __restrict__ nr = embed + nbase[t];
#pragma unroll
          for (int j = 0; j < 8; ++j) {
            float x0 = nr[off0 + j];
            _Float16 h0 = (_Float16)x0;
            ah[j]     = h0;
            al[j]     = (_Float16)(x0 - (float)h0);
            float x1 = nr[off0 + 8 + j];
            _Float16 h1 = (_Float16)x1;
            ah[8 + j] = h1;
            al[8 + j] = (_Float16)(x1 - (float)h1);
          }
        }
        acc[t] = __builtin_amdgcn_wmma_f32_16x16x32_f16(
                     false, ah, false, bh, (short)0, acc[t], false, false);
        acc[t] = __builtin_amdgcn_wmma_f32_16x16x32_f16(
                     false, ah, false, bl, (short)0, acc[t], false, false);
        acc[t] = __builtin_amdgcn_wmma_f32_16x16x32_f16(
                     false, al, false, bh, (short)0, acc[t], false, false);
      }
    }

    // Stable logsumexp over the 64 inner values. Each lane holds 32 of them
    // (8 per tile); every value is replicated on exactly 16 lanes.
    float m = -__builtin_inff();
#pragma unroll
    for (int t = 0; t < 4; ++t)
#pragma unroll
      for (int j = 0; j < 8; ++j) m = fmaxf(m, acc[t][j]);
    m = wave_max32(m);                                 // replication-safe

    float se = 0.0f;
#pragma unroll
    for (int t = 0; t < 4; ++t)
#pragma unroll
      for (int j = 0; j < 8; ++j) se += expf(acc[t][j] - m);
    se = wave_sum32(se) * 0.0625f;                     // undo 16x replication

    const float lse = m + logf(se);                    // logsumexp(a.n_k)
    const float z   = lse - s;                         // logsumexp(a.n_k - a.p)
    const float loss = (z > 0.0f) ? z + log1pf(expf(-z)) : log1pf(expf(z));
    local += loss;
  }

  __shared__ float red[8];
  if (lane == 0) red[waveInBlk] = local;
  __syncthreads();
  if (threadIdx.x == 0) {
    float t = 0.0f;
#pragma unroll
    for (int j = 0; j < 8; ++j) t += red[j];
    partial[blockIdx.x] = t;                           // deterministic partial
  }
}

// ---------------------------------------------------------------------------
// L2 term: sum of row norms over all 16384 embeddings. One wave per row.
// ---------------------------------------------------------------------------
__global__ __launch_bounds__(256) void l2_kernel(
    const float* __restrict__ embed, float* __restrict__ partial)
{
  const int lane = threadIdx.x & 31;
  const int w    = threadIdx.x >> 5;
  const int wave = blockIdx.x * 8 + w;                 // 0..511

  float local = 0.0f;
#pragma unroll 1
  for (int r = wave; r < B_TOTAL; r += L2_BLOCKS * 8) {
    float4 x = ((const float4*)(embed + (size_t)r * DIM))[lane];
    float ss = wave_sum32(x.x*x.x + x.y*x.y + x.z*x.z + x.w*x.w);
    local += sqrtf(ss);                                // wave-uniform
  }

  __shared__ float red[8];
  if (lane == 0) red[w] = local;
  __syncthreads();
  if (threadIdx.x == 0) {
    float t = 0.0f;
#pragma unroll
    for (int j = 0; j < 8; ++j) t += red[j];
    partial[blockIdx.x] = t;
  }
}

// ---------------------------------------------------------------------------
// Deterministic final reduction: single wave, fixed summation order.
// ---------------------------------------------------------------------------
__global__ __launch_bounds__(32) void finalize_kernel(
    const float* __restrict__ np_part,
    const float* __restrict__ l2_part,
    float*       __restrict__ out)
{
  const int lane = threadIdx.x;
  float s0 = 0.0f, s1 = 0.0f;
  for (int j = lane; j < NPAIR_BLOCKS; j += 32) s0 += np_part[j];
  for (int j = lane; j < L2_BLOCKS;    j += 32) s1 += l2_part[j];
  s0 = wave_sum32(s0);
  s1 = wave_sum32(s1);
  if (lane == 0)
    out[0] = s0 * (1.0f / (float)A_TOTAL) + 0.25f * (s1 * (1.0f / (float)B_TOTAL));
}

extern "C" void kernel_launch(void* const* d_in, const int* in_sizes, int n_in,
                              void* d_out, int out_size, void* d_ws, size_t ws_size,
                              hipStream_t stream) {
  (void)in_sizes; (void)n_in; (void)out_size;
  const float* embed   = (const float*)d_in[0];
  const int*   anc_ind = (const int*)  d_in[1];
  const int*   pos_ind = (const int*)  d_in[2];
  const int*   neg_ind = (const int*)  d_in[3];
  float*       out     = (float*)d_out;

  float* np_part = (float*)d_ws;               // [0, 512) floats
  float* l2_part = np_part + NPAIR_BLOCKS;     // [512, 576) floats

  // f16 hi/lo planes live after the partials, 256B-aligned.
  const size_t plane_off   = 4096;                          // bytes
  const size_t plane_elems = (size_t)B_TOTAL * DIM;         // 2M halves = 4MB
  const size_t need        = plane_off + 2 * plane_elems * sizeof(_Float16);

  if (ws_size >= need) {
    _Float16* hip_ = (_Float16*)((char*)d_ws + plane_off);
    _Float16* lop_ = hip_ + plane_elems;
    split_kernel<<<(B_TOTAL * DIM) / (256 * 4), 256, 0, stream>>>(embed, hip_, lop_);
    npair_kernel<true><<<NPAIR_BLOCKS, 256, 0, stream>>>(
        embed, hip_, lop_, anc_ind, pos_ind, neg_ind, np_part);
  } else {
    npair_kernel<false><<<NPAIR_BLOCKS, 256, 0, stream>>>(
        embed, (const _Float16*)nullptr, (const _Float16*)nullptr,
        anc_ind, pos_ind, neg_ind, np_part);
  }
  l2_kernel<<<L2_BLOCKS, 256, 0, stream>>>(embed, l2_part);
  finalize_kernel<<<1, 32, 0, stream>>>(np_part, l2_part, out);
}
